// RidgeletLayer_72670846649201
// MI455X (gfx1250) — matrix operand order repaired and checked by reading, not compiled, not run on hardware
//
#include <hip/hip_runtime.h>
#include <hip/hip_bf16.h>

typedef __attribute__((ext_vector_type(16))) __bf16 v16bf;
typedef __attribute__((ext_vector_type(8)))  float  v8f;

#define NB     16
#define IN_CH  3
#define OUT_CH 64
#define KS     7
#define HH_    512
#define WW_    512

// padded channel-summed input plane (3-halo baked in, zero borders)
#define PH        (HH_ + 6)         // 518 rows
#define PSTRIDE   528               // halves per padded row (dword friendly)
#define PLANE     (PH * PSTRIDE)    // halves per image  = 273504
#define A_BYTES   8192              // packed A fragments at d_ws offset 0

// per-workgroup output tile
#define TH   16
#define TW   32
#define LROWS (TH + 6)              // 22
#define LCOLS 40                    // LDS row stride in halves (>= TW+6)

// LDS half-offset for patch element K (K = kh*7+kw); K>=49 is A-zeroed padding
__device__ constexpr int kOff(int K) {
    return (K < 49) ? ((K / 7) * LCOLS + (K % 7)) : 0;
}

// ---------------------------------------------------------------------------
// Kernel 1: build ridgelet matrix in WMMA 16-bit A-fragment layout (bf16).
// apack[(mt*2+s)*32 + lane][h], h=0..15, Klocal = (h&7) + (lane>=16)*8 + (h>>3)*16
// ---------------------------------------------------------------------------
__global__ void build_a(const float* __restrict__ scales,
                        const float* __restrict__ dirs,
                        const float* __restrict__ pos,
                        void* apack_raw) {
    __bf16* apack = (__bf16*)apack_raw;
    const int tid    = threadIdx.x;         // 0..255
    const int lane   = tid & 31;
    const int frag   = tid >> 5;            // mt*2 + s
    const int mt     = frag >> 1;
    const int s      = frag & 1;
    const int laneLo = lane & 15;
    const int hiBit  = (lane >> 4) & 1;
    const int o      = mt * 16 + laneLo;    // out-channel

    const float cd  = cosf(dirs[o]);
    const float sd  = sinf(dirs[o]);
    const float p   = pos[o];
    const float scl = scales[o];

    #pragma unroll
    for (int h = 0; h < 16; ++h) {
        const int klocal = (h & 7) + hiBit * 8 + ((h >> 3) << 4);
        const int K = s * 32 + klocal;
        float v = 0.0f;
        if (K < 49) {
            const int kh = K / 7, kw = K % 7;
            const float tc = ((float)(kh - 3) * cd + (float)(kw - 3) * sd - p) / scl;
            const float t2 = tc * tc;
            v = expf(-0.5f * t2) - 0.5f * expf(-0.125f * t2);
        }
        apack[(tid << 4) + h] = (__bf16)v;
    }
}

// ---------------------------------------------------------------------------
// Kernel 2: xs[n, pr, pc] = sum over 3 input channels, with zeroed 3-halo.
// ---------------------------------------------------------------------------
__global__ void build_xs(const float* __restrict__ x, void* xs_raw) {
    __bf16* xs = (__bf16*)xs_raw;
    const unsigned idx = blockIdx.x * 256u + threadIdx.x;  // == n*PLANE + pr*PSTRIDE + pc
    const unsigned n   = idx / PLANE;
    const unsigned rem = idx - n * PLANE;
    const unsigned pr  = rem / PSTRIDE;
    const unsigned pc  = rem - pr * PSTRIDE;
    const int h = (int)pr - 3, w = (int)pc - 3;
    float v = 0.0f;
    if ((unsigned)h < HH_ && (unsigned)w < WW_) {
        const float* px = x + (size_t)n * IN_CH * HH_ * WW_ + (size_t)h * WW_ + w;
        v = px[0] + px[(size_t)HH_ * WW_] + px[2 * (size_t)HH_ * WW_];
    }
    xs[idx] = (__bf16)v;
}

// ---------------------------------------------------------------------------
// Kernel 3: implicit-GEMM conv via v_wmma_f32_16x16x32_bf16.
// Block = 8 waves; tile = 16x32 pixels x 64 channels for one image.
// ---------------------------------------------------------------------------
__global__ __launch_bounds__(256) void conv_wmma(const void* apack_raw,
                                                 const unsigned int* __restrict__ xs32,
                                                 float* __restrict__ out) {
    __shared__ unsigned int tile32[LROWS * (LCOLS / 2)];   // 22 x 20 dwords

    const int tid = threadIdx.x;
    const int n   = blockIdx.z;
    const int h0  = blockIdx.y * TH;
    const int w0  = blockIdx.x * TW;

    // ---- cooperative halo-tile load: 22 rows x 19 dwords (38 halves) ----
    const unsigned base_dw = (unsigned)n * (PLANE / 2)
                           + (unsigned)h0 * (PSTRIDE / 2)
                           + (unsigned)(w0 >> 1);
    for (int i = tid; i < LROWS * 19; i += 256) {
        const int row = i / 19;
        const int dw  = i - row * 19;
        tile32[row * (LCOLS / 2) + dw] =
            xs32[base_dw + (unsigned)row * (PSTRIDE / 2) + (unsigned)dw];
    }
    __syncthreads();
    const __bf16* tileH = (const __bf16*)tile32;

    const int  lane = tid & 31;
    const int  wv   = tid >> 5;        // wave id 0..7
    const bool hi   = lane >= 16;      // B/C layout: hi half-wave = upper K / M rows
    const int  j    = lane & 15;       // column within N-tile

    // ---- load A fragments (pre-packed to exact A-operand layout) ----
    const v16bf* apack = (const v16bf*)apack_raw;
    v16bf A[4][2];
    #pragma unroll
    for (int m = 0; m < 4; ++m)
        #pragma unroll
        for (int s = 0; s < 2; ++s)
            A[m][s] = apack[(m * 2 + s) * 32 + lane];

    // ---- each wave: 4 N-tiles of 16 pixels ----
    #pragma unroll
    for (int q = 0; q < 4; ++q) {
        const int nt   = wv * 4 + q;       // 0..31
        const int r    = nt >> 1;          // tile row 0..15
        const int cb   = (nt & 1) * 16;    // col base 0 or 16
        const int base = r * LCOLS + cb + j;

        // B fragments: element t of lane<16 holds K=Kbase+(t&7)+(t>>3)*16 of column j,
        //              lane>=16 holds K=Kbase+8+(t&7)+(t>>3)*16 of column j.
        v16bf blo, bhi;
        #pragma unroll
        for (int t = 0; t < 16; ++t) {
            const int o0 = kOff((t & 7) + ((t >> 3) << 4)), o1 = kOff(8 + (t & 7) + ((t >> 3) << 4));       // constants
            blo[t] = tileH[base + (hi ? o1 : o0)];           // cndmask + ds_load
        }
        #pragma unroll
        for (int t = 0; t < 16; ++t) {
            const int o0 = kOff(32 + (t & 7) + ((t >> 3) << 4)), o1 = kOff(40 + (t & 7) + ((t >> 3) << 4));
            bhi[t] = tileH[base + (hi ? o1 : o0)];
        }

        v8f acc[4] = {};
        #pragma unroll
        for (int m = 0; m < 4; ++m)
            acc[m] = __builtin_amdgcn_wmma_f32_16x16x32_bf16(
                false, A[m][0], false, blo, (short)0, acc[m], false, false);
        #pragma unroll
        for (int m = 0; m < 4; ++m)
            acc[m] = __builtin_amdgcn_wmma_f32_16x16x32_bf16(
                false, A[m][1], false, bhi, (short)0, acc[m], false, false);

        // ---- store: C layout lane<16 -> M=e, lane>=16 -> M=e+8; N=j ----
        const int hh = h0 + r;
        const int wc = w0 + cb + j;
        #pragma unroll
        for (int m = 0; m < 4; ++m) {
            #pragma unroll
            for (int e = 0; e < 8; ++e) {
                const int och = m * 16 + e + (hi ? 8 : 0);
                out[(((size_t)n * OUT_CH + och) * HH_ + hh) * WW_ + wc] = acc[m][e];
            }
        }
    }
}

// ---------------------------------------------------------------------------
extern "C" void kernel_launch(void* const* d_in, const int* in_sizes, int n_in,
                              void* d_out, int out_size, void* d_ws, size_t ws_size,
                              hipStream_t stream) {
    const float* x      = (const float*)d_in[0];
    const float* scales = (const float*)d_in[1];
    const float* dirs   = (const float*)d_in[2];
    const float* pos    = (const float*)d_in[3];

    void* apack = d_ws;                                   // 8 KB
    void* xs    = (void*)((char*)d_ws + A_BYTES);         // ~8.75 MB bf16 padded planes
    float* out  = (float*)d_out;

    build_a<<<1, 256, 0, stream>>>(scales, dirs, pos, apack);

    const unsigned total = NB * PLANE;                    // 4,376,064 (multiple of 256)
    build_xs<<<total / 256, 256, 0, stream>>>(x, xs);

    dim3 grid(WW_ / TW, HH_ / TH, NB);                    // 16 x 32 x 16
    conv_wmma<<<grid, 256, 0, stream>>>(apack, (const unsigned int*)xs, out);
}
